// PatchCrossAttentionFusion_86388972192222
// MI455X (gfx1250) — compile-verified, matrix operand order checked by reading, not run on hardware
//
#include <hip/hip_runtime.h>
#include <hip/hip_bf16.h>

// ---------------------------------------------------------------------------
// PatchCrossAttentionFusion for MI455X (gfx1250, wave32, WMMA)
//
// Per window (64 tokens x 256 ch): LN -> Q/K/V bf16 WMMA GEMMs -> per-head
// attention (1 wave = 1 head) -> out-proj WMMA GEMM + residual -> scatter.
// ---------------------------------------------------------------------------

typedef __attribute__((ext_vector_type(16))) __bf16 v16bf;
typedef __attribute__((ext_vector_type(8)))  float  v8f;

struct FragU { uint4 a; uint4 b; };   // 32 bytes == v16bf

__device__ __forceinline__ unsigned short f2bf(float x) {
    unsigned u = __builtin_bit_cast(unsigned, x);
    u += 0x7FFFu + ((u >> 16) & 1u);          // round-to-nearest-even
    return (unsigned short)(u >> 16);
}

// Load one 16x32 bf16 WMMA operand fragment.
// Layout per CDNA5 ISA 7.12.2 (16-bit A 16x32): lanes 0-15 -> rows, K 0..7 &
// 16..23; lanes 16-31 -> same rows, K 8..15 & 24..31.  The B (KxN) operand for
// a  C = X * W^T  GEMM uses the identical per-lane addressing on row-major W.
__device__ __forceinline__ v16bf load_frag(const unsigned short* base,
                                           int row0, int k0, int stride) {
    const int lane = threadIdx.x & 31;
    const int hlf  = lane >> 4;
    const int r    = lane & 15;
    const unsigned short* p = base + (row0 + r) * stride + k0 + hlf * 8;
    FragU f;
    f.a = *(const uint4*)(p);        // K offsets 0..7   (+8 for upper half)
    f.b = *(const uint4*)(p + 16);   // K offsets 16..23 (+8 for upper half)
    return __builtin_bit_cast(v16bf, f);
}

__device__ __forceinline__ v8f wmma_bf16(v16bf a, v16bf b, v8f c) {
    return __builtin_amdgcn_wmma_f32_16x16x32_bf16(
        false, a, false, b, (short)0, c, false, false);
}

// ---------------------------------------------------------------------------
// Pre-pass: convert the four 256x256 f32 weight matrices to bf16 in d_ws.
// ---------------------------------------------------------------------------
__global__ void cvt_weights(const float* __restrict__ Wq,
                            const float* __restrict__ Wk,
                            const float* __restrict__ Wv,
                            const float* __restrict__ Wo,
                            unsigned short* __restrict__ out) {
    int i = blockIdx.x * 256 + threadIdx.x;           // 0 .. 4*65536-1
    const float* src = (i < 65536) ? Wq : (i < 131072) ? Wk
                     : (i < 196608) ? Wv : Wo;
    out[i] = f2bf(src[i & 65535]);
}

// ---------------------------------------------------------------------------
// Fused window attention kernel. grid = 4096 windows, block = 256 (8 waves).
// ---------------------------------------------------------------------------
__global__ __launch_bounds__(256, 1)
void win_attn(const float* __restrict__ y_lf, const float* __restrict__ y_hf,
              const float* __restrict__ g_lf, const float* __restrict__ b_lf,
              const float* __restrict__ g_hf, const float* __restrict__ b_hf,
              const float* __restrict__ bq,   const float* __restrict__ bk,
              const float* __restrict__ bv,   const float* __restrict__ bo,
              const unsigned short* __restrict__ wbf,   // [Wq|Wk|Wv|Wo] bf16
              float* __restrict__ out) {
    // 256 KB LDS, regions reused across phases:
    //   [0      ,  64K) hfF   : hf tokens f32 (residual, LN src)   [persist]
    //   [64K    , 128K) lfF f32  ->  Qb (32K) + Kb (32K) bf16
    //   [128K   , 192K) lnlf+lnhf bf16 -> attn (8KB/wave) -> Ys f32 staging
    //   [192K   , 224K) hfB bf16 -> Ob bf16
    //   [224K   , 256K) Vt bf16 (256 d x 64 tok, transposed V)
    __shared__ char smem[262144];
    float*          hfF  = (float*)(smem);
    float*          lfF  = (float*)(smem + 65536);
    unsigned short* Qb   = (unsigned short*)(smem + 65536);
    unsigned short* Kb   = (unsigned short*)(smem + 65536 + 32768);
    unsigned short* lnlf = (unsigned short*)(smem + 131072);
    unsigned short* lnhf = (unsigned short*)(smem + 131072 + 32768);
    unsigned short* attnB= (unsigned short*)(smem + 131072);
    float*          Ys   = (float*)(smem + 131072);
    unsigned short* hfB  = (unsigned short*)(smem + 196608);
    unsigned short* Ob   = (unsigned short*)(smem + 196608);
    unsigned short* Vt   = (unsigned short*)(smem + 229376);

    const int w  = blockIdx.x;
    const int bb = w >> 10;
    const int wy = (w >> 5) & 31;
    const int wx = w & 31;
    const int h0 = wy * 8, w0 = wx * 8;

    const int tid  = threadIdx.x;
    const int wave = tid >> 5;
    const int lane = tid & 31;
    const int hlf  = lane >> 4;
    const int col  = lane & 15;

    // ---------------- Phase 1: window gather (coalesced b128 loads) --------
    {
        const int c = tid;   // one channel per thread
        const long base = (((long)(bb * 256 + c) * 256 + h0) * 256 + w0);
        const float* plf = y_lf + base;
        const float* phf = y_hf + base;
        #pragma unroll
        for (int py = 0; py < 8; ++py) {
            float4 a0 = *(const float4*)(plf + py * 256);
            float4 a1 = *(const float4*)(plf + py * 256 + 4);
            float4 c0 = *(const float4*)(phf + py * 256);
            float4 c1 = *(const float4*)(phf + py * 256 + 4);
            const int t0 = py * 8;
            float av[8] = {a0.x,a0.y,a0.z,a0.w,a1.x,a1.y,a1.z,a1.w};
            float cv[8] = {c0.x,c0.y,c0.z,c0.w,c1.x,c1.y,c1.z,c1.w};
            #pragma unroll
            for (int px = 0; px < 8; ++px) {
                lfF[(t0 + px) * 256 + c] = av[px];
                hfF[(t0 + px) * 256 + c] = cv[px];
                hfB[(t0 + px) * 256 + c] = f2bf(cv[px]);
            }
        }
    }
    __syncthreads();

    // ---------------- Phase 2: LayerNorm (4 lanes per token) ---------------
    {
        const int t  = tid >> 2;
        const int jq = tid & 3;
        #pragma unroll
        for (int g = 0; g < 2; ++g) {
            const float* src = (g == 0) ? lfF : hfF;
            const float* gg  = (g == 0) ? g_lf : g_hf;
            const float* bbp = (g == 0) ? b_lf : b_hf;
            unsigned short* dst = (g == 0) ? lnlf : lnhf;
            const float* row = src + t * 256 + jq * 64;
            float s = 0.f, ss = 0.f;
            #pragma unroll
            for (int i = 0; i < 16; ++i) {
                float4 v = *(const float4*)(row + i * 4);
                s  += v.x + v.y + v.z + v.w;
                ss += v.x*v.x + v.y*v.y + v.z*v.z + v.w*v.w;
            }
            s  += __shfl_xor(s, 1, 32);  s  += __shfl_xor(s, 2, 32);
            ss += __shfl_xor(ss, 1, 32); ss += __shfl_xor(ss, 2, 32);
            const float mean = s * (1.f / 256.f);
            const float var  = ss * (1.f / 256.f) - mean * mean;
            const float rstd = rsqrtf(var + 1e-5f);
            #pragma unroll 4
            for (int i = 0; i < 64; ++i) {
                const int c = jq * 64 + i;
                dst[t * 256 + c] =
                    f2bf((row[i] - mean) * rstd * gg[c] + bbp[c]);
            }
        }
    }
    __syncthreads();

    // ---------------- Phase 3: Q/K/V projection GEMMs (bf16 WMMA) ----------
    // 64x256x256 each; wave handles 2 N-tiles of 16 columns.
    for (int g = 0; g < 3; ++g) {
        const unsigned short* A  = (g == 0) ? lnlf : (g == 1) ? lnhf : hfB;
        const unsigned short* Wg = wbf + g * 65536;
        const float* bias        = (g == 0) ? bq : (g == 1) ? bk : bv;
        #pragma unroll
        for (int nt2 = 0; nt2 < 2; ++nt2) {
            const int n  = wave * 2 + nt2;
            const float bsc = bias[n * 16 + col];
            #pragma unroll
            for (int mt = 0; mt < 4; ++mt) {
                v8f acc = {0.f,0.f,0.f,0.f,0.f,0.f,0.f,0.f};
                #pragma unroll
                for (int k = 0; k < 8; ++k) {
                    v16bf a = load_frag(A,  mt * 16, k * 32, 256);
                    v16bf b = load_frag(Wg, n  * 16, k * 32, 256);
                    acc = wmma_bf16(a, b, acc);
                }
                if (g < 2) {
                    unsigned short* outp = (g == 0) ? Qb : Kb;
                    #pragma unroll
                    for (int j = 0; j < 8; ++j)
                        outp[(mt*16 + hlf*8 + j) * 256 + n*16 + col] =
                            f2bf(acc[j] + bsc);
                } else {  // V stored transposed: Vt[d][token]
                    #pragma unroll
                    for (int j = 0; j < 8; ++j)
                        Vt[(n*16 + col) * 64 + mt*16 + hlf*8 + j] =
                            f2bf(acc[j] + bsc);
                }
            }
        }
    }
    __syncthreads();

    // ---------------- Phase 4: attention, one wave per head ----------------
    {
        const int h = wave;
        unsigned short* attnW = attnB + h * 4096;      // 64x64 bf16
        const float scale = 0.17677669529663687f;      // 1/sqrt(32)
        #pragma unroll
        for (int mt = 0; mt < 4; ++mt) {
            v16bf aq = load_frag(Qb, mt * 16, h * 32, 256);
            v8f st[4];
            #pragma unroll
            for (int nt = 0; nt < 4; ++nt) {
                v16bf kb = load_frag(Kb, nt * 16, h * 32, 256);
                v8f z = {0.f,0.f,0.f,0.f,0.f,0.f,0.f,0.f};
                st[nt] = wmma_bf16(aq, kb, z);
            }
            // register softmax: row lives in one VGPR slot across 16 lanes
            #pragma unroll
            for (int j = 0; j < 8; ++j) {
                float x0 = st[0][j]*scale, x1 = st[1][j]*scale;
                float x2 = st[2][j]*scale, x3 = st[3][j]*scale;
                float m = fmaxf(fmaxf(x0, x1), fmaxf(x2, x3));
                m = fmaxf(m, __shfl_xor(m, 1, 32));
                m = fmaxf(m, __shfl_xor(m, 2, 32));
                m = fmaxf(m, __shfl_xor(m, 4, 32));
                m = fmaxf(m, __shfl_xor(m, 8, 32));
                float e0 = __expf(x0 - m), e1 = __expf(x1 - m);
                float e2 = __expf(x2 - m), e3 = __expf(x3 - m);
                float s = e0 + e1 + e2 + e3;
                s += __shfl_xor(s, 1, 32); s += __shfl_xor(s, 2, 32);
                s += __shfl_xor(s, 4, 32); s += __shfl_xor(s, 8, 32);
                const float inv = 1.0f / s;
                const int r = (mt*16 + hlf*8 + j) * 64 + col;
                attnW[r +  0] = f2bf(e0 * inv);
                attnW[r + 16] = f2bf(e1 * inv);
                attnW[r + 32] = f2bf(e2 * inv);
                attnW[r + 48] = f2bf(e3 * inv);
            }
        }
        __syncthreads();
        // O_h = attn (64x64) @ V_h (64x32) via Vt
        v16bf bv00 = load_frag(Vt, h*32 +  0,  0, 64);
        v16bf bv01 = load_frag(Vt, h*32 +  0, 32, 64);
        v16bf bv10 = load_frag(Vt, h*32 + 16,  0, 64);
        v16bf bv11 = load_frag(Vt, h*32 + 16, 32, 64);
        #pragma unroll
        for (int mt = 0; mt < 4; ++mt) {
            v16bf a0 = load_frag(attnW, mt * 16,  0, 64);
            v16bf a1 = load_frag(attnW, mt * 16, 32, 64);
            v8f z0 = {0.f,0.f,0.f,0.f,0.f,0.f,0.f,0.f};
            v8f o0 = wmma_bf16(a0, bv00, z0);
            o0     = wmma_bf16(a1, bv01, o0);
            v8f z1 = {0.f,0.f,0.f,0.f,0.f,0.f,0.f,0.f};
            v8f o1 = wmma_bf16(a0, bv10, z1);
            o1     = wmma_bf16(a1, bv11, o1);
            #pragma unroll
            for (int j = 0; j < 8; ++j) {
                const int row = mt*16 + hlf*8 + j;
                Ob[row * 256 + h*32 +      col] = f2bf(o0[j]);
                Ob[row * 256 + h*32 + 16 + col] = f2bf(o1[j]);
            }
        }
    }
    __syncthreads();

    // ---------------- Phase 5: out-proj GEMM + bias + residual -------------
    {
        const unsigned short* Wo = wbf + 3 * 65536;
        #pragma unroll
        for (int nt2 = 0; nt2 < 2; ++nt2) {
            const int n = wave * 2 + nt2;
            const float bsc = bo[n * 16 + col];
            #pragma unroll
            for (int mt = 0; mt < 4; ++mt) {
                v8f acc = {0.f,0.f,0.f,0.f,0.f,0.f,0.f,0.f};
                #pragma unroll
                for (int k = 0; k < 8; ++k) {
                    v16bf a = load_frag(Ob, mt * 16, k * 32, 256);
                    v16bf b = load_frag(Wo, n  * 16, k * 32, 256);
                    acc = wmma_bf16(a, b, acc);
                }
                #pragma unroll
                for (int j = 0; j < 8; ++j) {
                    const int row = mt*16 + hlf*8 + j;
                    const int cc  = n*16 + col;
                    Ys[row * 256 + cc] = acc[j] + bsc + hfF[row * 256 + cc];
                }
            }
        }
    }
    __syncthreads();

    // ---------------- Phase 6: window-reverse scatter (coalesced) ----------
    {
        const int c = tid;
        float* po = out + (((long)(bb * 256 + c) * 256 + h0) * 256 + w0);
        #pragma unroll
        for (int py = 0; py < 8; ++py) {
            const int t0 = py * 8;
            float4 r0, r1;
            r0.x = Ys[(t0+0)*256 + c]; r0.y = Ys[(t0+1)*256 + c];
            r0.z = Ys[(t0+2)*256 + c]; r0.w = Ys[(t0+3)*256 + c];
            r1.x = Ys[(t0+4)*256 + c]; r1.y = Ys[(t0+5)*256 + c];
            r1.z = Ys[(t0+6)*256 + c]; r1.w = Ys[(t0+7)*256 + c];
            *(float4*)(po + py * 256)     = r0;
            *(float4*)(po + py * 256 + 4) = r1;
        }
    }
}

extern "C" void kernel_launch(void* const* d_in, const int* in_sizes, int n_in,
                              void* d_out, int out_size, void* d_ws, size_t ws_size,
                              hipStream_t stream) {
    const float* y_lf = (const float*)d_in[0];
    const float* y_hf = (const float*)d_in[1];
    const float* g_lf = (const float*)d_in[2];
    const float* b_lf = (const float*)d_in[3];
    const float* g_hf = (const float*)d_in[4];
    const float* b_hf = (const float*)d_in[5];
    const float* Wq   = (const float*)d_in[6];
    const float* bq   = (const float*)d_in[7];
    const float* Wk   = (const float*)d_in[8];
    const float* bk   = (const float*)d_in[9];
    const float* Wv   = (const float*)d_in[10];
    const float* bv   = (const float*)d_in[11];
    const float* Wo   = (const float*)d_in[12];
    const float* bo   = (const float*)d_in[13];

    unsigned short* wbf = (unsigned short*)d_ws;      // 512 KB bf16 weights

    cvt_weights<<<1024, 256, 0, stream>>>(Wq, Wk, Wv, Wo, wbf);
    win_attn<<<4096, 256, 0, stream>>>(y_lf, y_hf, g_lf, b_lf, g_hf, b_hf,
                                       bq, bk, bv, bo, wbf, (float*)d_out);
}
